// RGCNBasisLayer_5446018531349
// MI455X (gfx1250) — matrix operand order repaired and checked by reading, not compiled, not run on hardware
//
#include <hip/hip_runtime.h>
#include <hip/hip_bf16.h>

typedef __attribute__((ext_vector_type(16))) _Float16 v16h;
typedef __attribute__((ext_vector_type(8)))  float    v8f;

#define N_NODES 60000
#define N_EDGES 600000
#define NREL    200
#define NBASIS  4
#define DIN     32
#define DOUT    32
#define DATT    32

// ---------------------------------------------------------------------------
// WMMA fragment layouts (CDNA5 ISA 7.12.2, 16-bit operands, wave32):
//   A (16x32, MxK): lane L -> m = L&15, h = L>>4; element j -> K = 8h + j (+8 if j>=8)
//   B (32x16, KxN): lane L -> n = L&15;          element j -> K = 8h + j (+8 if j>=8)
//   C/D (16x16 f32): lane L -> n = L&15 (+16 per N-half); VGPR g -> m = g + 8h
// Weights are staged into LDS pre-permuted into B-fragment order:
//   frag[(chunk*2 + nhalf)*32 + lane][j]  (16 contiguous halfs -> 1 v16h load)
// ---------------------------------------------------------------------------

// Stage a row-major [chunks*32 x 32] f32 weight matrix into B-fragment layout.
__device__ __forceinline__ void stage_b_frags(const float* __restrict__ w,
                                              v16h* __restrict__ lds,
                                              int chunks, int tid, int nthreads)
{
    _Float16* p = (_Float16*)lds;
    const int total = chunks * 2 * 32 * 16;
    for (int i = tid; i < total; i += nthreads) {
        const int j  = i & 15;
        const int ln = (i >> 4) & 31;
        const int nh = (i >> 9) & 1;
        const int c  = i >> 10;
        const int k  = 8 * (ln >> 4) + (j & 7) + ((j >> 3) << 4);        // K within chunk
        const int n  = (ln & 15) + 16 * nh;       // output column
        p[i] = (_Float16)w[(c * 32 + k) * 32 + n];
    }
}

// Load one lane's 16 A-fragment f32 values from a 128B-aligned 32-float row:
// K = [8h, 8h+8) ∪ [16+8h, 16+8h+8) -> two contiguous 32B runs (4x b128).
__device__ __forceinline__ void load_a16(const float* __restrict__ row, int h,
                                         float* __restrict__ v)
{
    const float4 r0 = *(const float4*)(row + 8 * h);
    const float4 r1 = *(const float4*)(row + 8 * h + 4);
    const float4 r2 = *(const float4*)(row + 8 * h + 16);
    const float4 r3 = *(const float4*)(row + 8 * h + 20);
    v[0]=r0.x; v[1]=r0.y; v[2]=r0.z;  v[3]=r0.w;
    v[4]=r1.x; v[5]=r1.y; v[6]=r1.z;  v[7]=r1.w;
    v[8]=r2.x; v[9]=r2.y; v[10]=r2.z; v[11]=r2.w;
    v[12]=r3.x; v[13]=r3.y; v[14]=r3.z; v[15]=r3.w;
}

__device__ __forceinline__ v16h to_f16(const float* __restrict__ v, float scale)
{
    v16h a;
#pragma unroll
    for (int j = 0; j < 16; ++j) a[j] = (_Float16)(scale * v[j]);
    return a;
}

// ===========================================================================
// Kernel 1: self-loop init   out[v,:] = x[v,:] @ self_loop_w + bias   (f32)
// ===========================================================================
__global__ __launch_bounds__(128)
void rgcn_selfloop_kernel(const float* __restrict__ x,
                          const float* __restrict__ slw,
                          const float* __restrict__ bias,
                          float* __restrict__ out)
{
    __shared__ v16h sWF[2 * 32];                  // 1 chunk x 2 nhalf x 32 lanes
    stage_b_frags(slw, sWF, 1, threadIdx.x, blockDim.x);
    __syncthreads();

    const int wave = threadIdx.x >> 5;
    const int lane = threadIdx.x & 31;
    const int tile = blockIdx.x * 4 + wave;       // 16 nodes / tile
    if (tile * 16 >= N_NODES) return;             // wave-uniform exit
    const int base = tile * 16;

    const int m  = lane & 15;
    const int h  = lane >> 4;
    const int n0 = lane & 15;

    float av[16];
    load_a16(x + (size_t)(base + m) * DIN, h, av);
    const v16h a  = to_f16(av, 1.f);
    const v16h b0 = sWF[lane];
    const v16h b1 = sWF[32 + lane];

    v8f c0 = {}, c1 = {};
    c0 = __builtin_amdgcn_wmma_f32_16x16x32_f16(false, a, false, b0, (short)0, c0, false, false);
    c1 = __builtin_amdgcn_wmma_f32_16x16x32_f16(false, a, false, b1, (short)0, c1, false, false);

    const float bb0 = bias[n0], bb1 = bias[n0 + 16];
#pragma unroll
    for (int g = 0; g < 8; ++g) {
        const int row = base + g + 8 * h;
        out[(size_t)row * DOUT + n0]      = c0[g] + bb0;
        out[(size_t)row * DOUT + n0 + 16] = c1[g] + bb1;
    }
}

// ===========================================================================
// Kernel 2: per-edge messages + attention, scatter-add into out.
// 16 edges per wave, 4 waves per block, 18 WMMAs per wave-tile.
// ===========================================================================
__global__ __launch_bounds__(128)
void rgcn_edge_kernel(const float* __restrict__ x,
                      const int*   __restrict__ src,
                      const int*   __restrict__ dst,
                      const int*   __restrict__ etype,
                      const int*   __restrict__ egraph,
                      const int*   __restrict__ tlabel,
                      const float* __restrict__ norm,
                      const float* __restrict__ rel_emds,
                      const float* __restrict__ basis,
                      const float* __restrict__ w_comp,
                      const float* __restrict__ A1w, const float* __restrict__ A1b,
                      const float* __restrict__ B1w, const float* __restrict__ B1b,
                      const float* __restrict__ A2w, const float* __restrict__ A2b,
                      const float* __restrict__ B2w, const float* __restrict__ B2b,
                      float* __restrict__ out)
{
    // Weights staged in B-fragment layout: 18KB LDS total.
    __shared__ v16h sBasisF[NBASIS * 2 * 32];     // basis  [128 x 32], 4 chunks
    __shared__ v16h sA1wF[3 * 2 * 32];            // A1_w   [ 96 x 32], 3 chunks
    __shared__ v16h sA2wF[2 * 2 * 32];            // A2_w   [ 64 x 32], 2 chunks
    stage_b_frags(basis, sBasisF, NBASIS, threadIdx.x, blockDim.x);
    stage_b_frags(A1w,   sA1wF,   3,      threadIdx.x, blockDim.x);
    stage_b_frags(A2w,   sA2wF,   2,      threadIdx.x, blockDim.x);
    __syncthreads();

    const int wave = threadIdx.x >> 5;
    const int lane = threadIdx.x & 31;
    const int tile = blockIdx.x * 4 + wave;       // 16 edges / tile
    if (tile * 16 >= N_EDGES) return;             // wave-uniform exit
    const int e0 = tile * 16;

    const int m  = lane & 15;                     // A-fragment row owned by lane
    const int h  = lane >> 4;
    const int n0 = lane & 15;                     // B-fragment / C-layout column

    // per-lane edge scalars for row m (L2-resident, 16-way shared)
    const int   sv  = src[e0 + m];
    const int   dv  = dst[e0 + m];
    const int   et  = etype[e0 + m];
    const int   gg  = egraph[e0 + m];
    const float nrm = norm[e0 + m];
    const int   tl  = tlabel[gg];

    const float* xs = x + (size_t)sv * DIN;
    const float* xd = x + (size_t)dv * DIN;
    const float* er = rel_emds + (size_t)(gg * NREL + et) * DATT;
    const float* tr = rel_emds + (size_t)(gg * NREL + tl) * DATT;

    float xsv[16], xdv[16], erv[16], trv[16];
    load_a16(xs, h, xsv);
    load_a16(xd, h, xdv);
    load_a16(er, h, erv);
    load_a16(tr, h, trv);

    // ---- message GEMM: [16 x 128] (coeff*norm-scaled x_src) @ basis[128 x 32]
    v8f macc0 = {}, macc1 = {};
#pragma unroll
    for (int c = 0; c < NBASIS; ++c) {
        const float cn = nrm * w_comp[et * NBASIS + c];
        const v16h a  = to_f16(xsv, cn);
        const v16h b0 = sBasisF[(c * 2 + 0) * 32 + lane];
        const v16h b1 = sBasisF[(c * 2 + 1) * 32 + lane];
        macc0 = __builtin_amdgcn_wmma_f32_16x16x32_f16(false, a, false, b0, (short)0, macc0, false, false);
        macc1 = __builtin_amdgcn_wmma_f32_16x16x32_f16(false, a, false, b1, (short)0, macc1, false, false);
    }

    // ---- attention MLP 1: [x_src | x_dst | edge_rel] (96) @ A1_w -> z1 (16x32)
    v8f z1a = {}, z1b = {};
#pragma unroll
    for (int c = 0; c < 3; ++c) {
        const float* ap = (c == 0) ? xsv : (c == 1) ? xdv : erv;
        const v16h a  = to_f16(ap, 1.f);
        const v16h b0 = sA1wF[(c * 2 + 0) * 32 + lane];
        const v16h b1 = sA1wF[(c * 2 + 1) * 32 + lane];
        z1a = __builtin_amdgcn_wmma_f32_16x16x32_f16(false, a, false, b0, (short)0, z1a, false, false);
        z1b = __builtin_amdgcn_wmma_f32_16x16x32_f16(false, a, false, b1, (short)0, z1b, false, false);
    }

    // ---- attention MLP 2: [edge_rel | edge_rel + tgt_rel] (64) @ A2_w -> z2
    v8f z2a = {}, z2b = {};
#pragma unroll
    for (int c = 0; c < 2; ++c) {
        v16h a;
        if (c == 0) {
            a = to_f16(erv, 1.f);
        } else {
#pragma unroll
            for (int j = 0; j < 16; ++j) a[j] = (_Float16)(erv[j] + trv[j]);
        }
        const v16h b0 = sA2wF[(c * 2 + 0) * 32 + lane];
        const v16h b1 = sA2wF[(c * 2 + 1) * 32 + lane];
        z2a = __builtin_amdgcn_wmma_f32_16x16x32_f16(false, a, false, b0, (short)0, z2a, false, false);
        z2b = __builtin_amdgcn_wmma_f32_16x16x32_f16(false, a, false, b1, (short)0, z2b, false, false);
    }

    // ---- alpha = sigmoid(B1(relu(z1))) + sigmoid(B2(relu(z2)))
    // C layout: VGPR g <-> row m = g + 8h; reduce over the 16 lanes of each half.
    const float a1b0 = A1b[n0], a1b1 = A1b[n0 + 16];
    const float b1w0 = B1w[n0], b1w1 = B1w[n0 + 16];
    const float a2b0 = A2b[n0], a2b1 = A2b[n0 + 16];
    const float b2w0 = B2w[n0], b2w1 = B2w[n0 + 16];
    const float b1bb = B1b[0],  b2bb = B2b[0];

    float alpha[8];
#pragma unroll
    for (int g = 0; g < 8; ++g) {
        float t1 = fmaxf(z1a[g] + a1b0, 0.f) * b1w0 + fmaxf(z1b[g] + a1b1, 0.f) * b1w1;
        float t2 = fmaxf(z2a[g] + a2b0, 0.f) * b2w0 + fmaxf(z2b[g] + a2b1, 0.f) * b2w1;
#pragma unroll
        for (int msk = 1; msk < 16; msk <<= 1) {  // stays within 16-lane half
            t1 += __shfl_xor(t1, msk, 32);
            t2 += __shfl_xor(t2, msk, 32);
        }
        const float s1 = 1.f / (1.f + __expf(-(t1 + b1bb)));
        const float s2 = 1.f / (1.f + __expf(-(t2 + b2bb)));
        alpha[g] = s1 + s2;   // alpha for row m = g + 8h, replicated across half
    }

    // ---- scatter: out[dst[e], n] += alpha_e * msg[e, n]
    // dst indices fetched lane->lane via shuffle (lane 'row' holds dst[e0+row]).
#pragma unroll
    for (int g = 0; g < 8; ++g) {
        const int row = g + 8 * h;                // edge index within tile
        const int dd  = __shfl(dv, row, 32);
        float* op = out + (size_t)dd * DOUT;
        atomicAdd(op + n0,      alpha[g] * macc0[g]);
        atomicAdd(op + n0 + 16, alpha[g] * macc1[g]);
    }
}

// ===========================================================================
// Kernel 3: in-place ReLU on out (vectorized, N*32 divisible by 4)
// ===========================================================================
__global__ void rgcn_relu_kernel(float4* __restrict__ out, int n4)
{
    const int i = blockIdx.x * blockDim.x + threadIdx.x;
    if (i < n4) {
        float4 v = out[i];
        v.x = fmaxf(v.x, 0.f);
        v.y = fmaxf(v.y, 0.f);
        v.z = fmaxf(v.z, 0.f);
        v.w = fmaxf(v.w, 0.f);
        out[i] = v;
    }
}

// ===========================================================================
extern "C" void kernel_launch(void* const* d_in, const int* in_sizes, int n_in,
                              void* d_out, int out_size, void* d_ws, size_t ws_size,
                              hipStream_t stream)
{
    (void)in_sizes; (void)n_in; (void)out_size; (void)d_ws; (void)ws_size;

    const float* x        = (const float*)d_in[0];
    const int*   src      = (const int*)  d_in[1];
    const int*   dst      = (const int*)  d_in[2];
    const int*   etype    = (const int*)  d_in[3];
    const int*   egraph   = (const int*)  d_in[4];
    const int*   tlabel   = (const int*)  d_in[5];
    const float* norm     = (const float*)d_in[6];
    const float* rel_emds = (const float*)d_in[7];
    const float* basis    = (const float*)d_in[8];
    const float* w_comp   = (const float*)d_in[9];
    const float* slw      = (const float*)d_in[10];
    const float* A1w      = (const float*)d_in[11];
    const float* A1b      = (const float*)d_in[12];
    const float* B1w      = (const float*)d_in[13];
    const float* B1b      = (const float*)d_in[14];
    const float* A2w      = (const float*)d_in[15];
    const float* A2b      = (const float*)d_in[16];
    const float* B2w      = (const float*)d_in[17];
    const float* B2b      = (const float*)d_in[18];
    const float* bias     = (const float*)d_in[19];
    float* out = (float*)d_out;

    // 1) init out = x @ self_loop_w + bias   (16 nodes/wave, 4 waves/block)
    const int nodeTiles  = N_NODES / 16;                 // 3750
    const int nodeBlocks = (nodeTiles + 3) / 4;          // 938
    rgcn_selfloop_kernel<<<nodeBlocks, 128, 0, stream>>>(x, slw, bias, out);

    // 2) edge messages + attention, atomic scatter into out
    const int edgeTiles  = N_EDGES / 16;                 // 37500
    const int edgeBlocks = (edgeTiles + 3) / 4;          // 9375
    rgcn_edge_kernel<<<edgeBlocks, 128, 0, stream>>>(
        x, src, dst, etype, egraph, tlabel, norm, rel_emds, basis, w_comp,
        A1w, A1b, B1w, B1b, A2w, A2b, B2w, B2b, out);

    // 3) ReLU
    const int n4 = (N_NODES * DOUT) / 4;
    rgcn_relu_kernel<<<(n4 + 255) / 256, 256, 0, stream>>>((float4*)out, n4);
}
